// MultiHeadAttention_56873956933829
// MI455X (gfx1250) — matrix operand order repaired and checked by reading, not compiled, not run on hardware
//
#include <hip/hip_runtime.h>

// ---------------------------------------------------------------------------
// MI455X (gfx1250) multi-head attention, bf16 WMMA pipeline with
// async global->LDS staging (double-buffered) on the GEMM core.
//   B=2, SQ=SKV=2048, D=1024, H=16, HD=64
// d_out = [ out: B*SQ*D f32 | attn: B*H*SQ*SKV f32 ]
// ---------------------------------------------------------------------------

typedef __attribute__((ext_vector_type(16))) __bf16 v16bf;
typedef __attribute__((ext_vector_type(8)))  __bf16 v8bf;
typedef __attribute__((ext_vector_type(8)))  float  v8f;

#define B_  2
#define SQ_ 2048
#define SKV_ 2048
#define D_  1024
#define H_  16
#define HD_ 64

// ----------------------- CDNA5 async copy primitives -----------------------
// GLOBAL_LOAD_ASYNC_TO_LDS_B128: per-lane 16B copy global -> LDS, ASYNCcnt.
__device__ __forceinline__ void async_copy16(unsigned lds_addr, const void* gptr) {
  asm volatile("global_load_async_to_lds_b128 %0, %1, off"
               :: "v"(lds_addr), "v"((unsigned long long)gptr)
               : "memory");
}
__device__ __forceinline__ void wait_async0() {
  asm volatile("s_wait_asynccnt 0x0" ::: "memory");
}
__device__ __forceinline__ unsigned lds_off(const void* p) {
  // generic LDS address: low 32 bits are the LDS byte offset
  return (unsigned)(unsigned long long)p;
}

// ----------------------------- f32 -> bf16 convert -------------------------
__global__ __launch_bounds__(256) void f32_to_bf16_kernel(
    const float* __restrict__ in, __bf16* __restrict__ out, int n4) {
  int i = blockIdx.x * blockDim.x + threadIdx.x;
  if (i >= n4) return;
  float4 v = reinterpret_cast<const float4*>(in)[i];
  alignas(8) __bf16 t[4];
  t[0] = (__bf16)v.x; t[1] = (__bf16)v.y; t[2] = (__bf16)v.z; t[3] = (__bf16)v.w;
  reinterpret_cast<uint2*>(out)[i] = *reinterpret_cast<uint2*>(t);
}

// --------------------- f32 [rows,cols] -> bf16 transpose -------------------
__global__ __launch_bounds__(256) void transpose_w_kernel(
    const float* __restrict__ W, __bf16* __restrict__ WT, int rows, int cols) {
  __shared__ float tile[32][33];
  int bc = blockIdx.x * 32;   // column tile of W
  int br = blockIdx.y * 32;   // row tile of W
  int tx = threadIdx.x & 31;
  int ty = threadIdx.x >> 5;  // 0..7
#pragma unroll
  for (int r = ty; r < 32; r += 8)
    tile[r][tx] = W[(size_t)(br + r) * cols + (bc + tx)];
  __syncthreads();
#pragma unroll
  for (int r = ty; r < 32; r += 8)
    WT[(size_t)(bc + r) * rows + (br + tx)] = (__bf16)tile[tx][r];
}

// ------------------------------ WMMA GEMM ----------------------------------
// C[m,n] = scale * sum_k A[m,k] * BT[n,k] + bias[n]
// 256 threads = 8 waves; block tile 128x64; wave tile 32x32 (2x2 WMMA 16x16).
// Double-buffered LDS; bf16 A/B staged with async global->LDS b128 copies.
// OMODE: 0 = f32 C[m*ldc+n], 1 = bf16 C[m*ldc+n], 2 = bf16 C[n*ldc+m]
#define BM 128
#define BN 64
#define BK 32
#define LDSP (BK + 8)   // 40 halves = 80B row stride (16B aligned)

template <typename AT>
__device__ __forceinline__ void stage_tile(
    const AT* __restrict__ A, int lda, const __bf16* __restrict__ BT, int ldb,
    int tileM, int tileN, int k0, int t,
    __bf16 (*Al)[LDSP], __bf16 (*Bl)[LDSP]) {
  // ---- A tile (128 x 32): 16 elems/thread ----
  const int arow = t >> 1;
  const int acol = (t & 1) * 16;
  if constexpr (sizeof(AT) == 2) {
    const __bf16* ag = (const __bf16*)A + (size_t)(tileM + arow) * lda + (k0 + acol);
    unsigned la = lds_off(&Al[arow][acol]);
    async_copy16(la,      ag);
    async_copy16(la + 16, ag + 8);
  } else {
    // f32 source (attention probabilities): load, convert, store synchronously
    const float* ag = (const float*)A + (size_t)(tileM + arow) * lda + (k0 + acol);
    alignas(16) __bf16 tmp[16];
#pragma unroll
    for (int q = 0; q < 4; ++q) {
      float4 f = reinterpret_cast<const float4*>(ag)[q];
      tmp[q * 4 + 0] = (__bf16)f.x; tmp[q * 4 + 1] = (__bf16)f.y;
      tmp[q * 4 + 2] = (__bf16)f.z; tmp[q * 4 + 3] = (__bf16)f.w;
    }
    *reinterpret_cast<uint4*>(&Al[arow][acol])     = *reinterpret_cast<uint4*>(&tmp[0]);
    *reinterpret_cast<uint4*>(&Al[arow][acol + 8]) = *reinterpret_cast<uint4*>(&tmp[8]);
  }
  // ---- BT tile (64 x 32): 8 elems/thread, async ----
  const int brow = t >> 2;
  const int bcol = (t & 3) * 8;
  const __bf16* bg = BT + (size_t)(tileN + brow) * ldb + (k0 + bcol);
  async_copy16(lds_off(&Bl[brow][bcol]), bg);
}

template <typename AT, int OMODE>
__global__ __launch_bounds__(256) void gemm_bf16_wmma(
    const AT* __restrict__ A, long long sAb, long long sAh, int lda,
    const __bf16* __restrict__ BT, long long sBb, long long sBh, int ldb,
    void* __restrict__ Cv, long long sCb, long long sCh, int ldc,
    const float* __restrict__ bias, float scale,
    int M, int N, int K, int Hper) {
  __shared__ __bf16 Al[2][BM][LDSP];
  __shared__ __bf16 Bl[2][BN][LDSP];

  const int bh = blockIdx.z;
  const int b = bh / Hper;
  const int h = bh - b * Hper;
  A  += (size_t)b * sAb + (size_t)h * sAh;
  BT += (size_t)b * sBb + (size_t)h * sBh;
  const size_t cOff = (size_t)b * sCb + (size_t)h * sCh;

  const int tileM = blockIdx.y * BM;
  const int tileN = blockIdx.x * BN;

  const int t    = threadIdx.x;
  const int wave = t >> 5;
  const int lane = t & 31;
  const int wr   = (wave >> 1) * 32;   // 4 wave-rows: 0,32,64,96
  const int wc   = (wave & 1) * 32;    // 2 wave-cols: 0,32
  const int half = lane >> 4;
  const int ml   = lane & 15;

  v8f acc[2][2] = {};

  const int nk = K / BK;
  stage_tile<AT>(A, lda, BT, ldb, tileM, tileN, 0, t, Al[0], Bl[0]);
  wait_async0();
  __syncthreads();

  for (int kc = 0; kc < nk; ++kc) {
    const int cur = kc & 1;
    if (kc + 1 < nk)
      stage_tile<AT>(A, lda, BT, ldb, tileM, tileN, (kc + 1) * BK, t,
                     Al[cur ^ 1], Bl[cur ^ 1]);

    // ---- fragments (CDNA5 WMMA VGPR layouts) ----
    v16bf afr[2], bfr[2];
#pragma unroll
    for (int i = 0; i < 2; ++i) {
      const int row = wr + i * 16 + ml;
      v8bf lo = *reinterpret_cast<const v8bf*>(&Al[cur][row][half * 8]);       // K0-7 / K8-15
      v8bf hi = *reinterpret_cast<const v8bf*>(&Al[cur][row][16 + half * 8]);  // K16-23 / K24-31
#pragma unroll
      for (int e = 0; e < 8; ++e) { afr[i][e] = lo[e]; afr[i][e + 8] = hi[e]; }
    }
#pragma unroll
    for (int j = 0; j < 2; ++j) {
      const int n = wc + j * 16 + ml;
      const int kb = half * 8;                                                 // K0-7,16-23 / K8-15,24-31
      v8bf lo = *reinterpret_cast<const v8bf*>(&Bl[cur][n][kb]);
      v8bf hi = *reinterpret_cast<const v8bf*>(&Bl[cur][n][16 + kb]);
#pragma unroll
      for (int e = 0; e < 8; ++e) { bfr[j][e] = lo[e]; bfr[j][e + 8] = hi[e]; }
    }
#pragma unroll
    for (int i = 0; i < 2; ++i)
#pragma unroll
      for (int j = 0; j < 2; ++j)
        acc[i][j] = __builtin_amdgcn_wmma_f32_16x16x32_bf16(
            false, afr[i], false, bfr[j], (short)0, acc[i][j], false, false);

    wait_async0();
    __syncthreads();
  }

  // ---- epilogue ----
#pragma unroll
  for (int i = 0; i < 2; ++i) {
#pragma unroll
    for (int j = 0; j < 2; ++j) {
      const int m0 = tileM + wr + i * 16;
      const int n  = tileN + wc + j * 16 + ml;
      const float bv = bias ? bias[n] : 0.0f;
#pragma unroll
      for (int r = 0; r < 8; ++r) {
        const int m = m0 + half * 8 + r;
        const float v = acc[i][j][r] * scale + bv;
        if constexpr (OMODE == 0) {
          reinterpret_cast<float*>(Cv)[cOff + (size_t)m * ldc + n] = v;
        } else if constexpr (OMODE == 1) {
          reinterpret_cast<__bf16*>(Cv)[cOff + (size_t)m * ldc + n] = (__bf16)v;
        } else {
          reinterpret_cast<__bf16*>(Cv)[cOff + (size_t)n * ldc + m] = (__bf16)v;
        }
      }
    }
  }
}

// ------------------------------ row softmax --------------------------------
__device__ inline float wave_max(float x) {
#pragma unroll
  for (int o = 16; o > 0; o >>= 1) x = fmaxf(x, __shfl_xor(x, o, 32));
  return x;
}
__device__ inline float wave_sum(float x) {
#pragma unroll
  for (int o = 16; o > 0; o >>= 1) x += __shfl_xor(x, o, 32);
  return x;
}

__global__ __launch_bounds__(256) void softmax_rows_kernel(float* __restrict__ attn) {
  __shared__ float sred[8];
  float* p = attn + (size_t)blockIdx.x * SKV_;
  const int t = threadIdx.x;
  float4 v0 = reinterpret_cast<float4*>(p)[t];
  float4 v1 = reinterpret_cast<float4*>(p)[t + 256];

  float mx = fmaxf(fmaxf(fmaxf(v0.x, v0.y), fmaxf(v0.z, v0.w)),
                   fmaxf(fmaxf(v1.x, v1.y), fmaxf(v1.z, v1.w)));
  mx = wave_max(mx);
  if ((t & 31) == 0) sred[t >> 5] = mx;
  __syncthreads();
  float m = sred[t & 7];
#pragma unroll
  for (int o = 4; o > 0; o >>= 1) m = fmaxf(m, __shfl_xor(m, o, 32));

  v0.x = __expf(v0.x - m); v0.y = __expf(v0.y - m);
  v0.z = __expf(v0.z - m); v0.w = __expf(v0.w - m);
  v1.x = __expf(v1.x - m); v1.y = __expf(v1.y - m);
  v1.z = __expf(v1.z - m); v1.w = __expf(v1.w - m);

  float s = (v0.x + v0.y + v0.z + v0.w) + (v1.x + v1.y + v1.z + v1.w);
  s = wave_sum(s);
  __syncthreads();
  if ((t & 31) == 0) sred[t >> 5] = s;
  __syncthreads();
  float tot = sred[t & 7];
#pragma unroll
  for (int o = 4; o > 0; o >>= 1) tot += __shfl_xor(tot, o, 32);
  const float inv = __frcp_rn(tot);

  v0.x *= inv; v0.y *= inv; v0.z *= inv; v0.w *= inv;
  v1.x *= inv; v1.y *= inv; v1.z *= inv; v1.w *= inv;
  reinterpret_cast<float4*>(p)[t]       = v0;
  reinterpret_cast<float4*>(p)[t + 256] = v1;
}

// ------------------------------- launch ------------------------------------
extern "C" void kernel_launch(void* const* d_in, const int* in_sizes, int n_in,
                              void* d_out, int out_size, void* d_ws, size_t ws_size,
                              hipStream_t stream) {
  const float* xq  = (const float*)d_in[0];
  const float* xkv = (const float*)d_in[1];
  const float* Wq  = (const float*)d_in[2];
  const float* bq  = (const float*)d_in[3];
  const float* Wk  = (const float*)d_in[4];
  const float* bk  = (const float*)d_in[5];
  const float* Wv  = (const float*)d_in[6];
  const float* bv  = (const float*)d_in[7];
  const float* Wo  = (const float*)d_in[8];
  const float* bo  = (const float*)d_in[9];

  float* out  = (float*)d_out;
  float* attn = out + (size_t)B_ * SQ_ * D_;

  // workspace carve (bytes)
  char* ws = (char*)d_ws;
  const size_t SZ_X = (size_t)B_ * SQ_ * D_ * 2;   // 8 MB each (SQ==SKV)
  const size_t SZ_W = (size_t)D_ * D_ * 2;         // 2 MB each
  __bf16* Xq  = (__bf16*)(ws);                 ws += SZ_X;
  __bf16* Xkv = (__bf16*)(ws);                 ws += SZ_X;
  __bf16* WqT = (__bf16*)(ws);                 ws += SZ_W;
  __bf16* WkT = (__bf16*)(ws);                 ws += SZ_W;
  __bf16* WvT = (__bf16*)(ws);                 ws += SZ_W;
  __bf16* WoT = (__bf16*)(ws);                 ws += SZ_W;
  __bf16* Qb  = (__bf16*)(ws);                 ws += SZ_X;
  __bf16* Kb  = (__bf16*)(ws);                 ws += SZ_X;
  __bf16* VT  = (__bf16*)(ws);                 ws += SZ_X;   // [B][D][SKV]
  __bf16* CTX = (__bf16*)(ws);                 ws += SZ_X;   // [B*SQ][D]

  const int nX = B_ * SQ_ * D_;  // 4M elements
  // 1) convert inputs to bf16
  f32_to_bf16_kernel<<<nX / 1024, 256, 0, stream>>>(xq,  Xq,  nX / 4);
  f32_to_bf16_kernel<<<nX / 1024, 256, 0, stream>>>(xkv, Xkv, nX / 4);
  // 2) transpose weights to bf16 (BT layout: [n][k])
  dim3 tg(D_ / 32, D_ / 32);
  transpose_w_kernel<<<tg, 256, 0, stream>>>(Wq, WqT, D_, D_);
  transpose_w_kernel<<<tg, 256, 0, stream>>>(Wk, WkT, D_, D_);
  transpose_w_kernel<<<tg, 256, 0, stream>>>(Wv, WvT, D_, D_);
  transpose_w_kernel<<<tg, 256, 0, stream>>>(Wo, WoT, D_, D_);

  // 3) Q/K projections: [B*SQ, D] x [D, D] -> bf16
  dim3 gp(D_ / BN, (B_ * SQ_) / BM, 1);
  gemm_bf16_wmma<__bf16, 1><<<gp, 256, 0, stream>>>(
      Xq, 0, 0, D_, WqT, 0, 0, D_, Qb, 0, 0, D_, bq, 1.0f, B_ * SQ_, D_, D_, 1);
  gemm_bf16_wmma<__bf16, 1><<<gp, 256, 0, stream>>>(
      Xkv, 0, 0, D_, WkT, 0, 0, D_, Kb, 0, 0, D_, bk, 1.0f, B_ * SKV_, D_, D_, 1);
  // V projection with transposed store -> VT[b][hf][skv]
  dim3 gv(D_ / BN, SKV_ / BM, B_);
  gemm_bf16_wmma<__bf16, 2><<<gv, 256, 0, stream>>>(
      Xkv, (long long)SKV_ * D_, 0, D_, WvT, 0, 0, D_,
      VT, (long long)D_ * SKV_, 0, SKV_, bv, 1.0f, SKV_, D_, D_, 1);

  // 4) scores: per (b,h)  S = (Q_h * K_h^T) / 8   -> attn region (f32)
  dim3 gs(SKV_ / BN, SQ_ / BM, B_ * H_);
  gemm_bf16_wmma<__bf16, 0><<<gs, 256, 0, stream>>>(
      Qb, (long long)SQ_ * D_, HD_, D_,
      Kb, (long long)SKV_ * D_, HD_, D_,
      attn, (long long)H_ * SQ_ * SKV_, (long long)SQ_ * SKV_, SKV_,
      nullptr, 0.125f, SQ_, SKV_, HD_, H_);

  // 5) softmax over keys, in place
  softmax_rows_kernel<<<B_ * H_ * SQ_, 256, 0, stream>>>(attn);

  // 6) ctx = attn(f32, cvt to bf16 on the fly) @ V  -> bf16 [B*SQ][D]
  dim3 gc(HD_ / BN, SQ_ / BM, B_ * H_);
  gemm_bf16_wmma<float, 1><<<gc, 256, 0, stream>>>(
      attn, (long long)H_ * SQ_ * SKV_, (long long)SQ_ * SKV_, SKV_,
      VT, (long long)D_ * SKV_, (long long)HD_ * SKV_, SKV_,
      CTX, (long long)SQ_ * D_, HD_, D_,
      nullptr, 1.0f, SQ_, HD_, SKV_, H_);

  // 7) out = ctx @ Wo + bo  -> f32 d_out
  gemm_bf16_wmma<__bf16, 0><<<gp, 256, 0, stream>>>(
      CTX, 0, 0, D_, WoT, 0, 0, D_, out, 0, 0, D_, bo, 1.0f, B_ * SQ_, D_, D_, 1);
}
